// SpatioTemporalGNN_77051713290277
// MI455X (gfx1250) — matrix operand order repaired and checked by reading, not compiled, not run on hardware
//
#include <hip/hip_runtime.h>
#include <hip/hip_bf16.h>
#include <math.h>

// ---------------------------------------------------------------------------
// MI455X (gfx1250) implementation, round 2.
//  - Dense GEMMs: v_wmma_f32_16x16x32_bf16, f32 accum.
//  - B strips staged into LDS once per block with GLOBAL_LOAD_ASYNC_TO_LDS_B128
//    (ASYNCcnt + s_wait_asynccnt), rows padded 16B to kill bank conflicts.
//  - Each wave computes 4 column tiles per A fragment (1 a-frag : 4 wmma).
//  - Graph aggregation (bandwidth-bound, ~7 GB HBM): float4 gathers +
//    per-element f32 atomic adds; 2*I self-loop folded into the init pass.
//  - GRU with h0==0 collapses (gh == bhh, Whh unused): fused 3-gate WMMA
//    kernel writes straight into the decoder concat buffer.
// ---------------------------------------------------------------------------

typedef __bf16 bf16_t;
typedef __attribute__((ext_vector_type(16))) __bf16 v16bf;
typedef __attribute__((ext_vector_type(8)))  __bf16 v8bf;
typedef __attribute__((ext_vector_type(8)))  float  v8f;
typedef __attribute__((address_space(3)))    char   lds_char_t;

// ---- async 16B global->LDS copy (ASYNCcnt) --------------------------------
static __device__ __forceinline__ void async_cp16(const bf16_t* g, bf16_t* l) {
  // AS3 pointer value == LDS byte offset, which is what VDST wants.
  unsigned loff = (unsigned)(size_t)(lds_char_t*)l;
  asm volatile("global_load_async_to_lds_b128 %0, %1, off"
               :: "v"(loff), "v"(g) : "memory");
}
static __device__ __forceinline__ void async_wait_and_barrier() {
#if __has_builtin(__builtin_amdgcn_s_wait_asynccnt)
  __builtin_amdgcn_s_wait_asynccnt(0);
#else
  asm volatile("s_wait_asynccnt 0" ::: "memory");
#endif
  __syncthreads();
}

// ---- WMMA fragment loaders (CDNA5 16-bit layouts, wave32) -----------------
// A (16x32, MxK): lanes 0-15: row M=lane, K=k0+{0..7,16..23};
//                 lanes 16-31: row M=lane-16, K=k0+{8..15,24..31}.
static __device__ __forceinline__ v16bf load_a_frag(const bf16_t* A, int lda,
                                                    int row0, int k0, int lane) {
  int r  = row0 + (lane & 15);
  int kh = (lane >> 4) * 8;
  const bf16_t* p = A + (size_t)r * lda + k0 + kh;
  v8bf lo = *(const v8bf*)p;
  v8bf hi = *(const v8bf*)(p + 16);
  v16bf a;
#pragma unroll
  for (int i = 0; i < 8; ++i) { a[i] = lo[i]; a[8 + i] = hi[i]; }
  return a;
}
// B (32x16, KxN) from LDS strip Bs[rows][ldb] (row r = output column):
// lanes 0-15: column = colLoc+lane, K=k0+{0..7,16..23}; lanes 16-31: same col, K=k0+{8..15,24..31}.
static __device__ __forceinline__ v16bf lds_b_frag(const bf16_t* Bs, int ldb,
                                                   int colLoc, int k0, int lane) {
  const bf16_t* p = Bs + (size_t)(colLoc + (lane & 15)) * ldb + k0 + ((lane >> 4) << 3);
  v8bf lo = *(const v8bf*)p;
  v8bf hi = *(const v8bf*)(p + 16);
  v16bf b;
#pragma unroll
  for (int i = 0; i < 8; ++i) { b[i] = lo[i]; b[8 + i] = hi[i]; }
  return b;
}
#define WMMA_BF16(a, b, c) \
  __builtin_amdgcn_wmma_f32_16x16x32_bf16(false, (a), false, (b), (short)0, (c), false, false)

// ---- WMMA GEMM: out[N,Ncols] = act(A[N,K] @ Bt^T + bias) ------------------
// block (32,4); block covers 64 rows x 64 cols; wave ty owns rowTile
// blockIdx.x*4+ty (clamped) and all 4 col tiles of the staged 64-col strip.
__global__ void k_gemm_bf16(const bf16_t* __restrict__ A,
                            const bf16_t* __restrict__ Bt,
                            const float* __restrict__ bias,
                            float* __restrict__ outF, bf16_t* __restrict__ outB,
                            int K, int Ncols, int relu, int rowTiles) {
  extern __shared__ char smem[];
  bf16_t* Bs = (bf16_t*)smem;
  const int ldb = K + 8;                        // +16B pad per row: no bank conflicts
  int lane = threadIdx.x, ty = threadIdx.y;
  int tid  = ty * 32 + lane;
  int colBase = blockIdx.y * 64;

  // Stage 64 rows of Bt (one output-column strip) via async LDS copies.
  int cpr   = K >> 3;                           // 16B chunks per row
  int total = cpr << 6;                         // 64 rows; multiple of 128
  for (int c = tid; c < total; c += 128) {
    int r = c / cpr, kk = (c - r * cpr) << 3;
    async_cp16(Bt + (size_t)(colBase + r) * K + kk, Bs + (size_t)r * ldb + kk);
  }
  async_wait_and_barrier();

  int rowTile = blockIdx.x * 4 + ty;
  int rtc  = rowTile < rowTiles ? rowTile : rowTiles - 1;  // clamp (benign dup)
  int row0 = rtc * 16;
  v8f acc0 = {}, acc1 = {}, acc2 = {}, acc3 = {};
  for (int k0 = 0; k0 < K; k0 += 32) {
    v16bf a  = load_a_frag(A, K, row0, k0, lane);
    v16bf b0 = lds_b_frag(Bs, ldb, 0,  k0, lane);
    v16bf b1 = lds_b_frag(Bs, ldb, 16, k0, lane);
    v16bf b2 = lds_b_frag(Bs, ldb, 32, k0, lane);
    v16bf b3 = lds_b_frag(Bs, ldb, 48, k0, lane);
    acc0 = WMMA_BF16(a, b0, acc0);
    acc1 = WMMA_BF16(a, b1, acc1);
    acc2 = WMMA_BF16(a, b2, acc2);
    acc3 = WMMA_BF16(a, b3, acc3);
  }
  if (rowTile >= rowTiles) return;

  int rb = row0 + ((lane >> 4) << 3);
  int lc = lane & 15;
  v8f accs[4] = {acc0, acc1, acc2, acc3};
#pragma unroll
  for (int t = 0; t < 4; ++t) {
    int oc = colBase + t * 16 + lc;
    float bv = bias ? bias[oc] : 0.0f;
#pragma unroll
    for (int r = 0; r < 8; ++r) {
      float v = accs[t][r] + bv;
      if (relu) v = fmaxf(v, 0.0f);
      size_t off = (size_t)(rb + r) * Ncols + oc;
      if (outF) outF[off] = v;
      if (outB) outB[off] = (bf16_t)v;
    }
  }
}

// ---- Fused GRU step (h0==0): out = (1-z)*tanh(i_n + bih_n + r*bhh_n) ------
// gi = h @ Wih^T => bf16(Wih)[768,256] is already Bt. Stages the 48 Wih rows
// (16 cols x 3 gates) for this block's column group into LDS.
__global__ void k_gru(const bf16_t* __restrict__ A,     // [N,256] bf16
                      const bf16_t* __restrict__ Wih,   // [768,256] bf16
                      const float* __restrict__ bih, const float* __restrict__ bhh,
                      bf16_t* __restrict__ dcat, int rowTiles) {
  extern __shared__ char smem[];
  bf16_t* Bs = (bf16_t*)smem;
  const int ldb = 264;                          // 256 + 8 pad
  int lane = threadIdx.x, ty = threadIdx.y;
  int tid  = ty * 32 + lane;
  int colBase = blockIdx.y * 16;                // 0..240

  // 48 rows x 32 chunks = 1536 chunks; multiple of 128.
  for (int c = tid; c < 1536; c += 128) {
    int r = c >> 5, kk = (c & 31) << 3;
    int g = r >> 4, rl = r & 15;
    async_cp16(Wih + (size_t)(colBase + g * 256 + rl) * 256 + kk,
               Bs + (size_t)r * ldb + kk);
  }
  async_wait_and_barrier();

  int rowTile = blockIdx.x * 4 + ty;
  int rtc  = rowTile < rowTiles ? rowTile : rowTiles - 1;
  int row0 = rtc * 16;
  v8f ar = {}, az = {}, an = {};
  for (int k0 = 0; k0 < 256; k0 += 32) {
    v16bf a  = load_a_frag(A, 256, row0, k0, lane);
    v16bf br = lds_b_frag(Bs, ldb, 0,  k0, lane);
    v16bf bz = lds_b_frag(Bs, ldb, 16, k0, lane);
    v16bf bn = lds_b_frag(Bs, ldb, 32, k0, lane);
    ar = WMMA_BF16(a, br, ar);
    az = WMMA_BF16(a, bz, az);
    an = WMMA_BF16(a, bn, an);
  }
  if (rowTile >= rowTiles) return;

  int   oc  = colBase + (lane & 15);
  float cbr = bih[oc] + bhh[oc];
  float cbz = bih[256 + oc] + bhh[256 + oc];
  float cbi = bih[512 + oc];
  float cbh = bhh[512 + oc];
  int   rb  = row0 + ((lane >> 4) << 3);
#pragma unroll
  for (int r = 0; r < 8; ++r) {
    float rg = 1.0f / (1.0f + expf(-(ar[r] + cbr)));
    float zg = 1.0f / (1.0f + expf(-(az[r] + cbz)));
    float ng = tanhf(an[r] + cbi + rg * cbh);
    dcat[(size_t)(rb + r) * 288 + oc] = (bf16_t)((1.0f - zg) * ng);
  }
}

// ---- LayerNorm over 256 cols (adds GCN bias first), optional ReLU ---------
__global__ void k_ln(const float* __restrict__ agg, const float* __restrict__ bias,
                     const float* __restrict__ g, const float* __restrict__ be,
                     bf16_t* __restrict__ out, int relu) {
  int lane = threadIdx.x;
  int row  = blockIdx.x * 8 + threadIdx.y;
  const float* p = agg + (size_t)row * 256;
  float v[8], s = 0.0f;
#pragma unroll
  for (int i = 0; i < 8; ++i) { int c = lane * 8 + i; v[i] = p[c] + bias[c]; s += v[i]; }
#pragma unroll
  for (int o = 16; o > 0; o >>= 1) s += __shfl_xor(s, o, 32);
  float mu = s * (1.0f / 256.0f);
  float vs = 0.0f;
#pragma unroll
  for (int i = 0; i < 8; ++i) { float d = v[i] - mu; vs += d * d; }
#pragma unroll
  for (int o = 16; o > 0; o >>= 1) vs += __shfl_xor(vs, o, 32);
  float is = rsqrtf(vs * (1.0f / 256.0f) + 1e-5f);
#pragma unroll
  for (int i = 0; i < 8; ++i) {
    int c = lane * 8 + i;
    float o = (v[i] - mu) * is * g[c] + be[c];
    if (relu) o = fmaxf(o, 0.0f);
    out[(size_t)row * 256 + c] = (bf16_t)o;
  }
}

// ---- Degree / normalization ----------------------------------------------
__global__ void k_deg_init(float* deg, int N) {
  int i = blockIdx.x * blockDim.x + threadIdx.x;
  if (i < N) deg[i] = 2.0f;                           // improved=True self loop
}
__global__ void k_deg_edges(const int* __restrict__ ei, const float* __restrict__ ew,
                            float* __restrict__ deg, int E) {
  int e = blockIdx.x * blockDim.x + threadIdx.x;
  if (e < E)
    __hip_atomic_fetch_add(&deg[ei[E + e]], ew[e], __ATOMIC_RELAXED,
                           __HIP_MEMORY_SCOPE_AGENT);
}
__global__ void k_dis(float* deg, int N) {
  int i = blockIdx.x * blockDim.x + threadIdx.x;
  if (i < N) deg[i] = rsqrtf(fmaxf(deg[i], 1e-12f));
}

// ---- Aggregation: init = zero + self-loop (2*dis^2), then edge scatter ----
__global__ void k_agg_init(const float* __restrict__ xw, const float* __restrict__ dis,
                           float* __restrict__ out, int N) {
  size_t i = (size_t)blockIdx.x * blockDim.x + threadIdx.x;   // N*64 float4 slots
  if (i >= (size_t)N * 64) return;
  int n = (int)(i >> 6);
  float s = dis[n]; s = 2.0f * s * s;
  float4 v = ((const float4*)xw)[i];
  float4 o = {v.x * s, v.y * s, v.z * s, v.w * s};
  ((float4*)out)[i] = o;
}
__global__ void k_agg_edges(const int* __restrict__ ei, const float* __restrict__ ew,
                            const float* __restrict__ dis, const float* __restrict__ xw,
                            float* __restrict__ out, int E) {
  int t = threadIdx.x & 63;                      // 64 threads * float4 = 256 cols
  int e = blockIdx.x * 4 + (threadIdx.x >> 6);   // 4 edges per 256-thread block
  if (e >= E) return;
  int s = ei[e], d = ei[E + e];
  float nrm = dis[s] * ew[e] * dis[d];
  float4 v = ((const float4*)(xw + (size_t)s * 256))[t];
  float* op = out + (size_t)d * 256 + t * 4;
  __hip_atomic_fetch_add(op + 0, v.x * nrm, __ATOMIC_RELAXED, __HIP_MEMORY_SCOPE_AGENT);
  __hip_atomic_fetch_add(op + 1, v.y * nrm, __ATOMIC_RELAXED, __HIP_MEMORY_SCOPE_AGENT);
  __hip_atomic_fetch_add(op + 2, v.z * nrm, __ATOMIC_RELAXED, __HIP_MEMORY_SCOPE_AGENT);
  __hip_atomic_fetch_add(op + 3, v.w * nrm, __ATOMIC_RELAXED, __HIP_MEMORY_SCOPE_AGENT);
}

// ---- Weight prep: convert / transpose to bf16 with K-padding --------------
__global__ void k_cvt_pad(const float* __restrict__ in, bf16_t* __restrict__ out,
                          int R, int C, int Cpad) {
  size_t i = (size_t)blockIdx.x * blockDim.x + threadIdx.x;
  if (i >= (size_t)R * Cpad) return;
  int r = (int)(i / Cpad), c = (int)(i % Cpad);
  out[i] = (c < C) ? (bf16_t)in[(size_t)r * C + c] : (bf16_t)0.0f;
}
__global__ void k_trans_pad(const float* __restrict__ in,   // [K,C] f32
                            bf16_t* __restrict__ out,       // [C,Kpad] bf16
                            int K, int C, int Kpad) {
  size_t i = (size_t)blockIdx.x * blockDim.x + threadIdx.x;
  if (i >= (size_t)C * Kpad) return;
  int c = (int)(i / Kpad), k = (int)(i % Kpad);
  out[i] = (k < K) ? (bf16_t)in[(size_t)k * C + c] : (bf16_t)0.0f;
}

// ---- Concat buffer cols 256..287: x (16) + zero pad (16) ------------------
__global__ void k_dcat_x(const float* __restrict__ x, bf16_t* __restrict__ dcat, int N) {
  int i = blockIdx.x * blockDim.x + threadIdx.x;
  if (i >= N * 32) return;
  int n = i >> 5, j = i & 31;
  dcat[(size_t)n * 288 + 256 + j] = (j < 16) ? (bf16_t)x[n * 16 + j] : (bf16_t)0.0f;
}

// ---- Final decoder layer (DOUT=2): tiny, plain VALU -----------------------
__global__ void k_dec3(const bf16_t* __restrict__ d2, const float* __restrict__ Wd3,
                       const float* __restrict__ bd3, float* __restrict__ out, int N) {
  int n = blockIdx.x * blockDim.x + threadIdx.x;
  if (n >= N) return;
  const bf16_t* p = d2 + (size_t)n * 128;
  float a0 = 0.0f, a1 = 0.0f;
  for (int i = 0; i < 128; ++i) {
    float f = (float)p[i];
    a0 = fmaf(f, Wd3[2 * i], a0);
    a1 = fmaf(f, Wd3[2 * i + 1], a1);
  }
  a0 += bd3[0]; a1 += bd3[1];
  out[2 * n]     = fminf(fmaxf(a0, -5.0f), 5.0f);
  out[2 * n + 1] = fminf(fmaxf(a1, -5.0f), 5.0f);
}

// ---------------------------------------------------------------------------
extern "C" void kernel_launch(void* const* d_in, const int* in_sizes, int n_in,
                              void* d_out, int out_size, void* d_ws, size_t ws_size,
                              hipStream_t stream) {
  const float* x   = (const float*)d_in[0];
  const int*   ei  = (const int*)  d_in[1];   // [2,E] int32: row0=src, row1=dst
  const float* ew  = (const float*)d_in[2];
  const float* W1  = (const float*)d_in[3];
  const float* b1  = (const float*)d_in[4];
  const float* g1  = (const float*)d_in[5];
  const float* be1 = (const float*)d_in[6];
  const float* W2  = (const float*)d_in[7];
  const float* b2  = (const float*)d_in[8];
  const float* g2  = (const float*)d_in[9];
  const float* be2 = (const float*)d_in[10];
  const float* Wih = (const float*)d_in[11];
  /* Whh = d_in[12]: unused, h0 == 0 */
  const float* bih = (const float*)d_in[13];
  const float* bhh = (const float*)d_in[14];
  const float* Wd1 = (const float*)d_in[15];
  const float* bd1 = (const float*)d_in[16];
  const float* Wd2 = (const float*)d_in[17];
  const float* bd2 = (const float*)d_in[18];
  const float* Wd3 = (const float*)d_in[19];
  const float* bd3 = (const float*)d_in[20];

  const int N = in_sizes[0] / 16;   // DIN = 16
  const int E = in_sizes[2];

  // Workspace carve (256B aligned). Total ~174 MB for N=50k.
  char* w = (char*)d_ws;
  auto carve = [&](size_t bytes) -> char* {
    char* p = w; w += (bytes + 255) & ~(size_t)255; return p;
  };
  bf16_t* xbf   = (bf16_t*)carve((size_t)N * 32 * 2);    // x padded K=16->32
  bf16_t* W1t   = (bf16_t*)carve(256 * 32 * 2);
  bf16_t* W2t   = (bf16_t*)carve(256 * 256 * 2);
  bf16_t* Wihb  = (bf16_t*)carve(768 * 256 * 2);
  bf16_t* Wd1t  = (bf16_t*)carve(256 * 288 * 2);         // K 272 -> 288
  bf16_t* Wd2t  = (bf16_t*)carve(128 * 256 * 2);
  float*  dis   = (float*) carve((size_t)N * 4);
  float*  bufA  = (float*) carve((size_t)N * 256 * 4);   // GEMM outs (xw)
  float*  bufB  = (float*) carve((size_t)N * 256 * 4);   // aggregation target
  bf16_t* hbf   = (bf16_t*)carve((size_t)N * 256 * 2);   // LN out / d1 reuse
  bf16_t* dcat  = (bf16_t*)carve((size_t)N * 288 * 2);   // [gru, x, pad]
  bf16_t* d2bf  = (bf16_t*)carve((size_t)N * 128 * 2);

  const dim3 wb(32, 4);
  const int  rowTiles = N / 16;                  // 3125 (N multiple of 16)
  const unsigned gx   = (unsigned)((rowTiles + 3) / 4);
  auto blks = [](size_t n, int t) { return (unsigned)((n + t - 1) / t); };
  auto shmem = [](int K) { return (size_t)64 * (K + 8) * 2; };

  // 0) weight prep + degree normalization
  k_cvt_pad  <<<blks((size_t)N * 32, 256), 256, 0, stream>>>(x, xbf, N, 16, 32);
  k_trans_pad<<<blks(256 * 32,  256), 256, 0, stream>>>(W1,  W1t,  16,  256, 32);
  k_trans_pad<<<blks(256 * 256, 256), 256, 0, stream>>>(W2,  W2t,  256, 256, 256);
  k_cvt_pad  <<<blks(768 * 256, 256), 256, 0, stream>>>(Wih, Wihb, 768, 256, 256);
  k_trans_pad<<<blks(256 * 288, 256), 256, 0, stream>>>(Wd1, Wd1t, 272, 256, 288);
  k_trans_pad<<<blks(128 * 256, 256), 256, 0, stream>>>(Wd2, Wd2t, 256, 128, 256);
  k_deg_init <<<blks(N, 256), 256, 0, stream>>>(dis, N);
  k_deg_edges<<<blks(E, 256), 256, 0, stream>>>(ei, ew, dis, E);
  k_dis      <<<blks(N, 256), 256, 0, stream>>>(dis, N);

  // 1) GCN layer 1: xw = x@W1 ; agg ; LN(+b1)+ReLU -> hbf
  k_gemm_bf16<<<dim3(gx, 4), wb, shmem(32), stream>>>(xbf, W1t, nullptr, bufA, nullptr, 32, 256, 0, rowTiles);
  k_agg_init <<<blks((size_t)N * 64, 256), 256, 0, stream>>>(bufA, dis, bufB, N);
  k_agg_edges<<<blks(E, 4), 256, 0, stream>>>(ei, ew, dis, bufA, bufB, E);
  k_ln       <<<N / 8, dim3(32, 8), 0, stream>>>(bufB, b1, g1, be1, hbf, 1);

  // 2) GCN layer 2: xw = h@W2 ; agg ; LN(+b2) (no relu) -> hbf
  k_gemm_bf16<<<dim3(gx, 4), wb, shmem(256), stream>>>(hbf, W2t, nullptr, bufA, nullptr, 256, 256, 0, rowTiles);
  k_agg_init <<<blks((size_t)N * 64, 256), 256, 0, stream>>>(bufA, dis, bufB, N);
  k_agg_edges<<<blks(E, 4), 256, 0, stream>>>(ei, ew, dis, bufA, bufB, E);
  k_ln       <<<N / 8, dim3(32, 8), 0, stream>>>(bufB, b2, g2, be2, hbf, 0);

  // 3) GRU step (fused, h0==0) -> dcat[:,0:256] ; x -> dcat[:,256:288]
  k_gru      <<<dim3(gx, 16), wb, (size_t)48 * 264 * 2, stream>>>(hbf, Wihb, bih, bhh, dcat, rowTiles);
  k_dcat_x   <<<blks((size_t)N * 32, 256), 256, 0, stream>>>(x, dcat, N);

  // 4) decoder MLP
  k_gemm_bf16<<<dim3(gx, 4), wb, shmem(288), stream>>>(dcat, Wd1t, bd1, nullptr, hbf,  288, 256, 1, rowTiles);
  k_gemm_bf16<<<dim3(gx, 2), wb, shmem(256), stream>>>(hbf,  Wd2t, bd2, nullptr, d2bf, 256, 128, 1, rowTiles);
  k_dec3     <<<blks(N, 256), 256, 0, stream>>>(d2bf, Wd3, bd3, (float*)d_out, N);
}
